// Model_8289286881762
// MI455X (gfx1250) — matrix operand order repaired and checked by reading, not compiled, not run on hardware
//
#include <hip/hip_runtime.h>

// ---------------- model constants ----------------
#define BATCH   4
#define SEQ     512
#define DMODEL  512
#define NHEADS  8
#define DHEAD   64
#define DFF     2048
#define NVOCAB  32000
#define NLAYERS 6
#define NTOK    (BATCH * SEQ)          // 2048 tokens per stream
#define BH      (BATCH * NHEADS)       // 32 attention matrices

// ---------------- WMMA types ----------------
typedef __bf16 bf16_t;
typedef bf16_t v16bf __attribute__((ext_vector_type(16)));
typedef float  v8f   __attribute__((ext_vector_type(8)));

__device__ __forceinline__ unsigned short f2bf(float f) {
    union { float f; unsigned u; } v; v.f = f;
    unsigned r = v.u + 0x7FFFu + ((v.u >> 16) & 1u);   // round to nearest even
    return (unsigned short)(r >> 16);
}
__device__ __forceinline__ unsigned pack_bf2(float lo, float hi) {
    return (unsigned)f2bf(lo) | ((unsigned)f2bf(hi) << 16);
}

// ================= batched tiled WMMA GEMM =================
// C[M,N] = act( alpha * A[M,K] x B(+bias) ),  A fp32, B fp32, C fp32
// BT=0: B stored [K,N] (x @ W).  BT=1: B stored [N,K] (Q @ K^T).
// Batch z: offsets  (z/zi)*s?o + (z%zi)*s?i  applied to A/B/C.
//
// LDS layouts (u32 = bf16 K-pair, row stride 20 u32 for 16B-aligned b128
// fragment loads and conflict-friendly banking):
//   Ash[m][k2]  m=0..127, k2=0..15   (A tile 128x32)
//   Bsh[n][k2]  n=0..63,  k2=0..15   (B tile 32x64, pre-transposed pairs)
#define MT 128
#define NT 64
#define KT 32
#define ASTR 20   // u32 per A row
#define BSTR 20   // u32 per B row

template<bool BT, bool RELU>
__global__ __launch_bounds__(256) void gemm_kernel(
    const float* __restrict__ A, const float* __restrict__ B,
    float* __restrict__ C, const float* __restrict__ bias,
    int M, int N, int K, int lda, int ldb, int ldc,
    int zi, long long sAo, long long sAi, long long sBo, long long sBi,
    long long sCo, long long sCi, float alpha)
{
    __shared__ unsigned AshU[MT * ASTR];   // 10240 B
    __shared__ unsigned BshU[NT * BSTR];   //  5120 B

    const int tid  = threadIdx.x;
    const int wave = tid >> 5;
    const int lane = tid & 31;
    const int g    = lane >> 4;      // lane group 0/1
    const int l16  = lane & 15;
    const int wm   = wave >> 1;      // 0..3 : 32-row sub-block
    const int wn   = wave & 1;       // 0..1 : 32-col sub-block

    const int bz = blockIdx.z;
    const int zo = bz / zi, zn = bz % zi;
    A += zo * sAo + zn * sAi;
    B += zo * sBo + zn * sBi;
    C += zo * sCo + zn * sCi;

    const int n0 = blockIdx.x * NT;
    const int m0 = blockIdx.y * MT;

    v8f acc[2][2];
#pragma unroll
    for (int i = 0; i < 2; ++i)
#pragma unroll
        for (int j = 0; j < 2; ++j)
#pragma unroll
            for (int r = 0; r < 8; ++r) acc[i][j][r] = 0.0f;

    for (int k0 = 0; k0 < K; k0 += KT) {
        // ---- load A tile: 128x32 fp32 -> bf16 K-pairs in LDS ----
#pragma unroll
        for (int l = 0; l < 4; ++l) {
            int idx = tid + l * 256;              // 0..1023
            int row = idx >> 3;                   // 0..127
            int c4  = (idx & 7) * 4;              // K offset 0..28
            float4 f = *(const float4*)(A + (long long)(m0 + row) * lda + k0 + c4);
            uint2 p;
            p.x = pack_bf2(f.x, f.y);
            p.y = pack_bf2(f.z, f.w);
            *(uint2*)(AshU + row * ASTR + (c4 >> 1)) = p;
        }
        // ---- load B tile -> Bsh[n][k2] (bf16 K-pairs) ----
        if (!BT) {
            // B[K,N]: each thread reads rows k,k+1 x 4 cols, packs 4 K-pairs
            int k2 = tid >> 4;                    // 0..15
            int c4 = (tid & 15) * 4;              // n offset 0..60
            const float* b0 = B + (long long)(k0 + 2 * k2)     * ldb + n0 + c4;
            const float* b1 = B + (long long)(k0 + 2 * k2 + 1) * ldb + n0 + c4;
            float4 r0 = *(const float4*)b0;
            float4 r1 = *(const float4*)b1;
            BshU[(c4 + 0) * BSTR + k2] = pack_bf2(r0.x, r1.x);
            BshU[(c4 + 1) * BSTR + k2] = pack_bf2(r0.y, r1.y);
            BshU[(c4 + 2) * BSTR + k2] = pack_bf2(r0.z, r1.z);
            BshU[(c4 + 3) * BSTR + k2] = pack_bf2(r0.w, r1.w);
        } else {
            // B[N,K]: pairs already along K -> contiguous uint2 store
#pragma unroll
            for (int l = 0; l < 2; ++l) {
                int idx = tid + l * 256;          // 0..511
                int n   = idx >> 3;               // 0..63
                int c4  = (idx & 7) * 4;          // K offset 0..28
                float4 f = *(const float4*)(B + (long long)(n0 + n) * ldb + k0 + c4);
                uint2 p;
                p.x = pack_bf2(f.x, f.y);
                p.y = pack_bf2(f.z, f.w);
                *(uint2*)(BshU + n * BSTR + (c4 >> 1)) = p;
            }
        }
        __syncthreads();

        // ---- fragments: each = two aligned b128 LDS loads ----
        union { uint4 q[2]; unsigned u[8]; v16bf v; } af[2], bf_[2];
#pragma unroll
        for (int i = 0; i < 2; ++i) {
            const unsigned* Ar = AshU + (wm * 32 + i * 16 + l16) * ASTR;
            af[i].q[0] = *(const uint4*)(Ar + 4 * g);       // K = 8g+2j
            af[i].q[1] = *(const uint4*)(Ar + 8 + 4 * g);   // K = 16+8g+2j
        }
#pragma unroll
        for (int j = 0; j < 2; ++j) {
            const unsigned* Br = BshU + (wn * 32 + j * 16 + l16) * BSTR + 4 * g;
            bf_[j].q[0] = *(const uint4*)(Br);              // K = 8g+2j
            bf_[j].q[1] = *(const uint4*)(Br + 8);
        }
        // ---- 4 WMMAs per wave per K-chunk ----
#pragma unroll
        for (int i = 0; i < 2; ++i)
#pragma unroll
            for (int j = 0; j < 2; ++j)
                acc[i][j] = __builtin_amdgcn_wmma_f32_16x16x32_bf16(
                    false, af[i].v, false, bf_[j].v, (short)0, acc[i][j], false, false);
        __syncthreads();
    }

    // ---- epilogue: alpha, bias, relu ----
#pragma unroll
    for (int i = 0; i < 2; ++i)
#pragma unroll
    for (int j = 0; j < 2; ++j) {
        int col = n0 + wn * 32 + j * 16 + l16;
        float bv = bias ? bias[col] : 0.0f;
#pragma unroll
        for (int r = 0; r < 8; ++r) {
            int row = m0 + wm * 32 + i * 16 + g * 8 + r;
            float v = acc[i][j][r] * alpha + bv;
            if (RELU) v = fmaxf(v, 0.0f);
            C[(long long)row * ldc + col] = v;
        }
    }
}

// ================= softmax (one wave32 per 512-wide row) =================
__device__ __forceinline__ float wave_max(float v) {
#pragma unroll
    for (int o = 16; o > 0; o >>= 1) v = fmaxf(v, __shfl_xor(v, o, 32));
    return v;
}
__device__ __forceinline__ float wave_sum(float v) {
#pragma unroll
    for (int o = 16; o > 0; o >>= 1) v += __shfl_xor(v, o, 32);
    return v;
}

__global__ __launch_bounds__(128) void softmax_kernel(float* __restrict__ S, int causal)
{
    const int wave = threadIdx.x >> 5;
    const int lane = threadIdx.x & 31;
    const long long row = (long long)blockIdx.x * 4 + wave;
    float* p = S + row * SEQ;
    const int q = (int)(row % SEQ);
    const int limit = causal ? (q + 1) : SEQ;

    float v[16];
#pragma unroll
    for (int j = 0; j < 16; ++j) {
        int i = lane + j * 32;
        v[j] = (i < limit) ? p[i] : -1e9f;
    }
    float mx = v[0];
#pragma unroll
    for (int j = 1; j < 16; ++j) mx = fmaxf(mx, v[j]);
    mx = wave_max(mx);
    float s = 0.0f;
#pragma unroll
    for (int j = 0; j < 16; ++j) { v[j] = __expf(v[j] - mx); s += v[j]; }
    s = wave_sum(s);
    float inv = 1.0f / s;
#pragma unroll
    for (int j = 0; j < 16; ++j) p[lane + j * 32] = v[j] * inv;
}

// ================= fused residual add + LayerNorm =================
// out = g * (x+a - mean) / (sqrt(var_unbiased) + eps) + b   (matches reference)
__global__ __launch_bounds__(128) void add_ln_kernel(
    const float* __restrict__ x, const float* __restrict__ a,
    const float* __restrict__ gam, const float* __restrict__ bet,
    float* __restrict__ out)
{
    const int wave = threadIdx.x >> 5;
    const int lane = threadIdx.x & 31;
    const long long row = (long long)blockIdx.x * 4 + wave;
    const long long base = row * DMODEL;

    float v[16];
    float sum = 0.0f;
#pragma unroll
    for (int j = 0; j < 16; ++j) {
        int i = lane + j * 32;
        v[j] = x[base + i] + a[base + i];
        sum += v[j];
    }
    sum = wave_sum(sum);
    float mean = sum * (1.0f / DMODEL);
    float ss = 0.0f;
#pragma unroll
    for (int j = 0; j < 16; ++j) { float d = v[j] - mean; ss += d * d; }
    ss = wave_sum(ss);
    float var = ss * (1.0f / (DMODEL - 1));       // unbiased, like torch .std()
    float inv = 1.0f / (sqrtf(var) + 1e-6f);
#pragma unroll
    for (int j = 0; j < 16; ++j) {
        int i = lane + j * 32;
        out[base + i] = gam[i] * (v[j] - mean) * inv + bet[i];
    }
}

// ================= embedding + positional encoding =================
__global__ __launch_bounds__(128) void embed_kernel(
    const int* __restrict__ tok, const float* __restrict__ emb,
    float* __restrict__ out)
{
    const int t = blockIdx.x;                  // token index 0..NTOK-1
    const int pos = t % SEQ;
    const long long row = (long long)tok[t] * DMODEL;
    const float scale = 22.62741699796952f;    // sqrt(512)
#pragma unroll
    for (int j = 0; j < 4; ++j) {
        int d = threadIdx.x * 4 + j;
        int half = d >> 1;
        // den = 10000^(-(2*tmp)/DMODEL), tmp = 2*half  ->  10000^(-half/128)
        float den = __powf(10000.0f, -(float)half * (1.0f / 128.0f));
        float ang = (float)pos * den;
        float pe = (d & 1) ? __cosf(ang) : __sinf(ang);
        out[(long long)t * DMODEL + d] = emb[row + d] * scale + pe;
    }
}

// ================= host-side dispatch =================
static void gemm(hipStream_t st, const float* A, const float* B, const float* bias,
                 float* C, int M, int N, int K, int lda, int ldb, int ldc,
                 int nbatch, int zi,
                 long long sAo, long long sAi, long long sBo, long long sBi,
                 long long sCo, long long sCi,
                 float alpha, bool bt, bool relu)
{
    dim3 grid(N / NT, M / MT, nbatch), block(256);
    if (bt) {
        if (relu) hipLaunchKernelGGL((gemm_kernel<true,  true >), grid, block, 0, st, A,B,C,bias,M,N,K,lda,ldb,ldc,zi,sAo,sAi,sBo,sBi,sCo,sCi,alpha);
        else      hipLaunchKernelGGL((gemm_kernel<true,  false>), grid, block, 0, st, A,B,C,bias,M,N,K,lda,ldb,ldc,zi,sAo,sAi,sBo,sBi,sCo,sCi,alpha);
    } else {
        if (relu) hipLaunchKernelGGL((gemm_kernel<false, true >), grid, block, 0, st, A,B,C,bias,M,N,K,lda,ldb,ldc,zi,sAo,sAi,sBo,sBi,sCo,sCi,alpha);
        else      hipLaunchKernelGGL((gemm_kernel<false, false>), grid, block, 0, st, A,B,C,bias,M,N,K,lda,ldb,ldc,zi,sAo,sAi,sBo,sBi,sCo,sCi,alpha);
    }
}

// simple non-batched GEMM: C = A[M,K] @ W[K,N] + bias  (optional relu)
static void gemm_proj(hipStream_t st, const float* A, const float* W, const float* bias,
                      float* C, int M, int N, int K, bool relu)
{
    gemm(st, A, W, bias, C, M, N, K, K, N, N, 1, 1, 0,0,0,0,0,0, 1.0f, false, relu);
}

extern "C" void kernel_launch(void* const* d_in, const int* in_sizes, int n_in,
                              void* d_out, int out_size, void* d_ws, size_t ws_size,
                              hipStream_t stream)
{
    (void)in_sizes; (void)n_in; (void)out_size; (void)ws_size;

    // ---- inputs (setup_inputs dict order, nested dicts flattened in key order) ----
    const int*   x       = (const int*)  d_in[0];
    const int*   y       = (const int*)  d_in[1];
    const float* emb_src = (const float*)d_in[2];
    const float* emb_tgt = (const float*)d_in[3];
    // encoder params: base 4
    const float *eWq=(const float*)d_in[4],  *ebq=(const float*)d_in[5];
    const float *eWk=(const float*)d_in[6],  *ebk=(const float*)d_in[7];
    const float *eWv=(const float*)d_in[8],  *ebv=(const float*)d_in[9];
    const float *eWo=(const float*)d_in[10], *ebo=(const float*)d_in[11];
    const float *eW1=(const float*)d_in[12], *eb1=(const float*)d_in[13];
    const float *eW2=(const float*)d_in[14], *eb2=(const float*)d_in[15];
    const float *eg1=(const float*)d_in[16], *ebe1=(const float*)d_in[17];
    const float *eg2=(const float*)d_in[18], *ebe2=(const float*)d_in[19];
    // decoder params: base 20
    const float *sWq=(const float*)d_in[20], *sbq=(const float*)d_in[21];
    const float *sWk=(const float*)d_in[22], *sbk=(const float*)d_in[23];
    const float *sWv=(const float*)d_in[24], *sbv=(const float*)d_in[25];
    const float *sWo=(const float*)d_in[26], *sbo=(const float*)d_in[27];
    const float *cWq=(const float*)d_in[28], *cbq=(const float*)d_in[29];
    const float *cWk=(const float*)d_in[30], *cbk=(const float*)d_in[31];
    const float *cWv=(const float*)d_in[32], *cbv=(const float*)d_in[33];
    const float *cWo=(const float*)d_in[34], *cbo=(const float*)d_in[35];
    const float *dW1=(const float*)d_in[36], *db1=(const float*)d_in[37];
    const float *dW2=(const float*)d_in[38], *db2=(const float*)d_in[39];
    const float *dg1=(const float*)d_in[40], *dbe1=(const float*)d_in[41];
    const float *dg2=(const float*)d_in[42], *dbe2=(const float*)d_in[43];
    const float *dg3=(const float*)d_in[44], *dbe3=(const float*)d_in[45];
    const float *genW=(const float*)d_in[46], *genB=(const float*)d_in[47];

    float* logits = (float*)d_out;

    // ---- workspace layout (fp32) ----
    const long long TD = (long long)NTOK * DMODEL;           // 1,048,576
    float* ws  = (float*)d_ws;
    float* qb  = ws;                   // [NTOK, DMODEL]
    float* kb  = qb + TD;
    float* vb  = kb + TD;
    float* zb  = vb + TD;              // concat-head attention result
    float* ab  = zb + TD;              // attention / ffn output (residual add input)
    float* sc  = ab + TD;              // scores [BH, SEQ, SEQ] = 8,388,608
    float* ff  = sc + (long long)BH * SEQ * SEQ;   // [NTOK, DFF]
    float* hb  = ff + (long long)NTOK * DFF;       // encoder state -> memory
    float* gb  = hb + TD;              // decoder state

    const long long SD = (long long)SEQ * DMODEL;  // 262144
    const long long SS = (long long)SEQ * SEQ;     // 262144
    const float att_scale = 0.125f;                // 1/sqrt(64)

    dim3 lnGrid(NTOK / 4), lnBlk(128);
    dim3 smGrid(BH * SEQ / 4), smBlk(128);

    // helper: multi-head attention into `ab`
    auto run_mha = [&](const float* qin, const float* kin, const float* vin,
                       const float* Wq, const float* bq_, const float* Wk_, const float* bk_,
                       const float* Wv_, const float* bv_, const float* Wo_, const float* bo_,
                       int causal)
    {
        gemm_proj(stream, qin, Wq,  bq_, qb, NTOK, DMODEL, DMODEL, false);
        gemm_proj(stream, kin, Wk_, bk_, kb, NTOK, DMODEL, DMODEL, false);
        gemm_proj(stream, vin, Wv_, bv_, vb, NTOK, DMODEL, DMODEL, false);
        // scores[bh] = alpha * Q_bh @ K_bh^T   (BT GEMM, batched over BH)
        gemm(stream, qb, kb, nullptr, sc, SEQ, SEQ, DHEAD,
             DMODEL, DMODEL, SEQ, BH, NHEADS,
             SD, DHEAD, SD, DHEAD, (long long)NHEADS * SS, SS,
             att_scale, true, false);
        hipLaunchKernelGGL(softmax_kernel, smGrid, smBlk, 0, stream, sc, causal);
        // Z[bh] = attn_bh @ V_bh  -> written into concat-head layout
        gemm(stream, sc, vb, nullptr, zb, SEQ, DHEAD, SEQ,
             SEQ, DMODEL, DMODEL, BH, NHEADS,
             (long long)NHEADS * SS, SS, SD, DHEAD, SD, DHEAD,
             1.0f, false, false);
        gemm_proj(stream, zb, Wo_, bo_, ab, NTOK, DMODEL, DMODEL, false);
    };

    const long long DD = (long long)DMODEL * DMODEL;
    const long long DF = (long long)DMODEL * DFF;

    // ---- embeddings ----
    hipLaunchKernelGGL(embed_kernel, dim3(NTOK), dim3(128), 0, stream, x, emb_src, hb);
    hipLaunchKernelGGL(embed_kernel, dim3(NTOK), dim3(128), 0, stream, y, emb_tgt, gb);

    // ---- encoder ----
    for (int i = 0; i < NLAYERS; ++i) {
        run_mha(hb, hb, hb,
                eWq + i*DD, ebq + i*DMODEL, eWk + i*DD, ebk + i*DMODEL,
                eWv + i*DD, ebv + i*DMODEL, eWo + i*DD, ebo + i*DMODEL, 0);
        hipLaunchKernelGGL(add_ln_kernel, lnGrid, lnBlk, 0, stream,
                           hb, ab, eg1 + i*DMODEL, ebe1 + i*DMODEL, hb);
        gemm_proj(stream, hb, eW1 + i*DF, eb1 + (long long)i*DFF, ff, NTOK, DFF, DMODEL, true);
        gemm_proj(stream, ff, eW2 + i*DF, eb2 + i*DMODEL, ab, NTOK, DMODEL, DFF, false);
        hipLaunchKernelGGL(add_ln_kernel, lnGrid, lnBlk, 0, stream,
                           hb, ab, eg2 + i*DMODEL, ebe2 + i*DMODEL, hb);
    }
    // hb now holds encoder memory

    // ---- decoder ----
    for (int i = 0; i < NLAYERS; ++i) {
        run_mha(gb, gb, gb,
                sWq + i*DD, sbq + i*DMODEL, sWk + i*DD, sbk + i*DMODEL,
                sWv + i*DD, sbv + i*DMODEL, sWo + i*DD, sbo + i*DMODEL, 1);
        hipLaunchKernelGGL(add_ln_kernel, lnGrid, lnBlk, 0, stream,
                           gb, ab, dg1 + i*DMODEL, dbe1 + i*DMODEL, gb);
        run_mha(gb, hb, hb,
                cWq + i*DD, cbq + i*DMODEL, cWk + i*DD, cbk + i*DMODEL,
                cWv + i*DD, cbv + i*DMODEL, cWo + i*DD, cbo + i*DMODEL, 0);
        hipLaunchKernelGGL(add_ln_kernel, lnGrid, lnBlk, 0, stream,
                           gb, ab, dg2 + i*DMODEL, dbe2 + i*DMODEL, gb);
        gemm_proj(stream, gb, dW1 + i*DF, db1 + (long long)i*DFF, ff, NTOK, DFF, DMODEL, true);
        gemm_proj(stream, ff, dW2 + i*DF, db2 + i*DMODEL, ab, NTOK, DMODEL, DFF, false);
        hipLaunchKernelGGL(add_ln_kernel, lnGrid, lnBlk, 0, stream,
                           gb, ab, dg3 + i*DMODEL, dbe3 + i*DMODEL, gb);
    }

    // ---- generator: logits = g @ genW + genB  (2048 x 32000 x 512) ----
    gemm_proj(stream, gb, genW, genB, logits, NTOK, NVOCAB, DMODEL, false);
}
